// GCLBlock_17171279249861
// MI455X (gfx1250) — compile-verified, matrix operand order checked by reading, not run on hardware
//
#include <hip/hip_runtime.h>
#include <math.h>

typedef __attribute__((ext_vector_type(16))) _Float16 v16h;
typedef __attribute__((ext_vector_type(8)))  float    v8f;
typedef __attribute__((ext_vector_type(2)))  float    v2f;

#define HW96   96
#define NPOS   (96 * 96)     // 9216
#define CDIM   64
#define NHEADS 4
#define HDIM   16
#define KSZ    13
#define NKROW  13
#define LROW   420           // padded LDS row stride (floats); 416 logits + pad
#define NLOG   416           // 13 rows * 32 cols

// ---------------------------------------------------------------------------
// Pack f32 weight matrix B[K x N] (row-major) into WMMA-lane-ordered f16
// blocks: per (k-chunk of 32, n-tile of 16) a 512-elem block [lane*16 + j]:
//   j <  8 : K = kc*32 + half*8 + j
//   j >= 8 : K = kc*32 + 16 + half*8 + (j-8)
// so each lane's 16 B values are contiguous (two global_load_b128 per tile).
// ---------------------------------------------------------------------------
__global__ void pack_b_f16(const float* __restrict__ B,
                           _Float16* __restrict__ Bp, int N, int K) {
    int i = blockIdx.x * blockDim.x + threadIdx.x;
    if (i >= N * K) return;
    int j    = i & 15;
    int lane = (i >> 4) & 31;
    int tile = i >> 9;
    int ntn  = N >> 4;
    int kc   = tile / ntn;
    int nt   = tile - kc * ntn;
    int half = lane >> 4;
    int col  = lane & 15;
    int kk   = (j < 8) ? (half * 8 + j) : (16 + half * 8 + (j - 8));
    Bp[i] = (_Float16)B[(kc * 32 + kk) * N + nt * 16 + col];
}

// ---------------------------------------------------------------------------
// LayerNorm over C=64, one wave per row, 8 rows per 256-thread block.
// ---------------------------------------------------------------------------
__global__ void ln_kernel(const float* __restrict__ x,
                          const float* __restrict__ g,
                          const float* __restrict__ b,
                          _Float16* __restrict__ out, int nrows) {
    int wave = threadIdx.x >> 5;
    int lane = threadIdx.x & 31;
    int row  = blockIdx.x * (blockDim.x >> 5) + wave;
    if (row >= nrows) return;

    const float* xr = x + row * CDIM;
    float x0 = xr[2 * lane], x1 = xr[2 * lane + 1];

    float s = x0 + x1;
    #pragma unroll
    for (int off = 16; off > 0; off >>= 1) s += __shfl_xor(s, off, 32);
    float mu = s * (1.0f / 64.0f);

    float d0 = x0 - mu, d1 = x1 - mu;
    float v = d0 * d0 + d1 * d1;
    #pragma unroll
    for (int off = 16; off > 0; off >>= 1) v += __shfl_xor(v, off, 32);
    float rstd = rsqrtf(v * (1.0f / 64.0f) + 1e-5f);

    out[row * CDIM + 2 * lane]     = (_Float16)(d0 * rstd * g[2 * lane]     + b[2 * lane]);
    out[row * CDIM + 2 * lane + 1] = (_Float16)(d1 * rstd * g[2 * lane + 1] + b[2 * lane + 1]);
}

// ---------------------------------------------------------------------------
// WMMA GEMM, 16x64 output per wave (4 accumulators), packed B.
// ---------------------------------------------------------------------------
__global__ void gemm_wmma_f16(const _Float16* __restrict__ A,
                              const _Float16* __restrict__ Bp,
                              const float* __restrict__ bias,
                              const float* __restrict__ res,
                              float* __restrict__ outF,
                              _Float16* __restrict__ outH,
                              int M, int N, int K, int gelu) {
    int wid = blockIdx.x * (blockDim.x >> 5) + (threadIdx.x >> 5);
    int ngrp = N >> 6;
    int nwaves = (M >> 4) * ngrp;
    if (wid >= nwaves) return;                      // wave-uniform exit

    int mt = wid / ngrp;
    int ng = wid - mt * ngrp;
    int ntbase = ng * 4;
    int ntn = N >> 4;

    int lane = threadIdx.x & 31;
    int r16  = lane & 15;
    int half = lane >> 4;
    int arow = mt * 16 + r16;

    v8f acc0 = {}, acc1 = {}, acc2 = {}, acc3 = {};

    for (int kc = 0; kc < (K >> 5); ++kc) {
        v16h a;
        const _Float16* ap = A + arow * K + kc * 32 + half * 8;
        #pragma unroll
        for (int j = 0; j < 8; ++j) {
            a[j]     = ap[j];
            a[8 + j] = ap[16 + j];
        }
        const _Float16* bp = Bp + (size_t)(kc * ntn + ntbase) * 512 + lane * 16;
        v16h b0 = *(const v16h*)(bp);
        v16h b1 = *(const v16h*)(bp + 512);
        v16h b2 = *(const v16h*)(bp + 1024);
        v16h b3 = *(const v16h*)(bp + 1536);
        acc0 = __builtin_amdgcn_wmma_f32_16x16x32_f16(false, a, false, b0, (short)0, acc0, false, false);
        acc1 = __builtin_amdgcn_wmma_f32_16x16x32_f16(false, a, false, b1, (short)0, acc1, false, false);
        acc2 = __builtin_amdgcn_wmma_f32_16x16x32_f16(false, a, false, b2, (short)0, acc2, false, false);
        acc3 = __builtin_amdgcn_wmma_f32_16x16x32_f16(false, a, false, b3, (short)0, acc3, false, false);
    }

    v8f accs[4] = {acc0, acc1, acc2, acc3};
    #pragma unroll
    for (int t = 0; t < 4; ++t) {
        int col = (ntbase + t) * 16 + r16;
        float bv = bias ? bias[col] : 0.0f;
        #pragma unroll
        for (int r = 0; r < 8; ++r) {
            int row = mt * 16 + r + 8 * half;
            float v = accs[t][r] + bv;
            if (res)  v += res[row * N + col];
            if (gelu) v = 0.5f * v * (1.0f + erff(v * 0.70710678118f));
            if (outF) outF[row * N + col] = v;
            if (outH) outH[row * N + col] = (_Float16)v;
        }
    }
}

// ---------------------------------------------------------------------------
// WMMA dilated neighborhood attention.
// One wave handles (head, group(gh,gw), group-row iih, w-half wt): 16 queries.
// All 16 queries share key rows sh..sh+12 (sh depends only on iih); their
// 13-wide col windows all lie within the 32-col group row, so compute dense
// 16x32 logits per key row with fp32 WMMA (K=16 as 4x K=4 steps), with
// rpb bias + (-1e30) mask preloaded as the C operand. Softmax stats per query
// from LDS, then AV as a 16x416 x 416x16 fp32 WMMA chain with
// p = exp(l-max)/sum built on the fly as the A operand.
// ---------------------------------------------------------------------------
__global__ void __launch_bounds__(64)
na_attn_wmma(const float* __restrict__ qkv,
             const float* __restrict__ rpb,
             _Float16* __restrict__ out16) {
    __shared__ float lds[2 * 16 * LROW];           // 53.8 KB, 2 waves/block

    int wline = threadIdx.x >> 5;
    int wid   = blockIdx.x * 2 + wline;            // 0 .. 2303
    int wt    = wid & 1;
    int t     = wid >> 1;
    int iih   = t & 31;
    t >>= 5;
    int grp   = t % 9;
    int head  = t / 9;
    int gh = grp / 3, gw = grp % 3;

    int lane = threadIdx.x & 31;
    int r16  = lane & 15;
    int half = lane >> 4;

    int sh    = min(max(iih - 6, 0), 19);
    int relh0 = sh - iih + 12;

    float* L = lds + wline * (16 * LROW);

    // ---- Q tile as fp32 WMMA A-operand: 4 chunks of K=4 ----
    // A layout (16x4 f32): row m = lane&15, VGPR j holds K = half*2 + j.
    int qiiw = wt * 16 + r16;
    int qpos = (gh + 3 * iih) * HW96 + gw + 3 * qiiw;
    const float* qp = qkv + qpos * 192 + head * HDIM;
    v2f qa[4];
    #pragma unroll
    for (int c = 0; c < 4; ++c) {
        v2f qv = *(const v2f*)(qp + c * 4 + half * 2);
        qa[c][0] = qv[0] * 0.25f;                  // hd^-0.5
        qa[c][1] = qv[1] * 0.25f;
    }

    const float* rp = rpb + head * 625;

    // ---- logits: 13 key rows x 2 key-col tiles, bias+mask in C operand ----
    for (int rr = 0; rr < NKROW; ++rr) {
        int relh  = relh0 + rr;
        int kbase = (gh + 3 * (sh + rr)) * HW96 + gw;
        #pragma unroll
        for (int kt = 0; kt < 2; ++kt) {
            int skw = kt * 16 + r16;               // this lane's key col
            v8f acc;
            #pragma unroll
            for (int rL = 0; rL < 8; ++rL) {       // D rows: m = rL + 8*half
                int m   = rL + 8 * half;
                int iiw = wt * 16 + m;
                int sw  = min(max(iiw - 6, 0), 19);
                bool valid = (skw >= sw) && (skw <= sw + 12);
                int relw = min(max(skw - iiw + 12, 0), 24);   // clamped: safe read
                float bv = rp[relh * 25 + relw];
                acc[rL] = valid ? bv : -1e30f;
            }
            // B layout (4x16 f32): col = lane&15, VGPR j holds K = half*2 + j
            const float* kp = qkv + (kbase + 3 * skw) * 192 + 64 + head * HDIM;
            #pragma unroll
            for (int c = 0; c < 4; ++c) {
                v2f kb = *(const v2f*)(kp + c * 4 + half * 2);
                acc = __builtin_amdgcn_wmma_f32_16x16x4_f32(
                    false, qa[c], false, kb, (short)0, acc, false, false);
            }
            #pragma unroll
            for (int rL = 0; rL < 8; ++rL) {
                int m = rL + 8 * half;
                L[m * LROW + rr * 32 + kt * 16 + r16] = acc[rL];
            }
        }
    }

    // ---- per-query softmax stats (lane handles query lane&15; both halves
    //      compute redundantly so all 32 lanes hold their A-row's stats) ----
    const float* Lq = L + r16 * LROW;
    float mx = -1e30f;
    for (int i = 0; i < NLOG; ++i) mx = fmaxf(mx, Lq[i]);
    float ssum = 0.0f;
    for (int i = 0; i < NLOG; ++i) ssum += __expf(Lq[i] - mx);
    float inv = 1.0f / ssum;

    // ---- AV: out(16x16) = P(16x416) @ V(416x16), 104 fp32 wmmas ----
    v8f oacc = {};
    for (int kk = 0; kk < NLOG / 4; ++kk) {
        int kv0 = kk * 4 + half * 2;
        v2f pa;
        pa[0] = __expf(Lq[kv0]     - mx) * inv;    // A: row r16, K=kv0,kv0+1
        pa[1] = __expf(Lq[kv0 + 1] - mx) * inv;
        int rrA = kv0 >> 5,       swA = kv0 & 31;
        int rrB = (kv0 + 1) >> 5, swB = (kv0 + 1) & 31;
        const float* vA = qkv + ((gh + 3 * (sh + rrA)) * HW96 + gw + 3 * swA) * 192 + 128 + head * HDIM;
        const float* vB = qkv + ((gh + 3 * (sh + rrB)) * HW96 + gw + 3 * swB) * 192 + 128 + head * HDIM;
        v2f vb;
        vb[0] = vA[r16];                           // B: col d = r16
        vb[1] = vB[r16];
        oacc = __builtin_amdgcn_wmma_f32_16x16x4_f32(
            false, pa, false, vb, (short)0, oacc, false, false);
    }

    // ---- store output tile (f16 for proj GEMM): D(m, d=r16) ----
    #pragma unroll
    for (int rL = 0; rL < 8; ++rL) {
        int m    = rL + 8 * half;
        int opos = (gh + 3 * iih) * HW96 + gw + 3 * (wt * 16 + m);
        out16[opos * CDIM + head * HDIM + r16] = (_Float16)(oacc[rL]);
    }
}

// ---------------------------------------------------------------------------
// Host-side launcher
// ---------------------------------------------------------------------------
extern "C" void kernel_launch(void* const* d_in, const int* in_sizes, int n_in,
                              void* d_out, int out_size, void* d_ws, size_t ws_size,
                              hipStream_t stream) {
    const float* x       = (const float*)d_in[0];
    const float* norm1_g = (const float*)d_in[1];
    const float* norm1_b = (const float*)d_in[2];
    const float* qkv_w   = (const float*)d_in[3];
    const float* qkv_b   = (const float*)d_in[4];
    const float* rpb     = (const float*)d_in[5];
    const float* proj_w  = (const float*)d_in[6];
    const float* proj_b  = (const float*)d_in[7];
    const float* norm2_g = (const float*)d_in[8];
    const float* norm2_b = (const float*)d_in[9];
    const float* fc1_w   = (const float*)d_in[10];
    const float* fc1_b   = (const float*)d_in[11];
    const float* fc2_w   = (const float*)d_in[12];
    const float* fc2_b   = (const float*)d_in[13];

    _Float16* h16     = (_Float16*)d_ws;                        // 9216*64
    _Float16* qkvwP   = h16    + NPOS * CDIM;                   // 64*192 packed
    float*    qkvbuf  = (float*)(qkvwP + CDIM * 3 * CDIM);      // 9216*192 f32
    _Float16* attn16  = (_Float16*)(qkvbuf + NPOS * 3 * CDIM);  // 9216*64
    _Float16* projwP  = attn16 + NPOS * CDIM;                   // 64*64 packed
    float*    x2      = (float*)(projwP + CDIM * CDIM);         // 9216*64 f32
    _Float16* h2_16   = (_Float16*)(x2 + NPOS * CDIM);          // 9216*64
    _Float16* fc1wP   = h2_16  + NPOS * CDIM;                   // 64*256 packed
    _Float16* m16     = fc1wP  + CDIM * 256;                    // 9216*256
    _Float16* fc2wP   = m16    + NPOS * 256;                    // 256*64 packed

    // 1) pack weights (f32 -> WMMA-lane-ordered f16)
    pack_b_f16<<<(CDIM * 3 * CDIM + 255) / 256, 256, 0, stream>>>(qkv_w, qkvwP, 3 * CDIM, CDIM);
    pack_b_f16<<<(CDIM * CDIM    + 255) / 256, 256, 0, stream>>>(proj_w, projwP, CDIM, CDIM);
    pack_b_f16<<<(CDIM * 256     + 255) / 256, 256, 0, stream>>>(fc1_w, fc1wP, 256, CDIM);
    pack_b_f16<<<(256 * CDIM     + 255) / 256, 256, 0, stream>>>(fc2_w, fc2wP, CDIM, 256);

    // 2) LN1
    ln_kernel<<<NPOS / 8, 256, 0, stream>>>(x, norm1_g, norm1_b, h16, NPOS);

    // 3) QKV GEMM: (9216x64)@(64x192) -> f32 qkvbuf   (1728 waves)
    {
        int waves = (NPOS / 16) * (3 * CDIM / 64);
        gemm_wmma_f16<<<(waves + 7) / 8, 256, 0, stream>>>(
            h16, qkvwP, qkv_b, nullptr, qkvbuf, nullptr, NPOS, 3 * CDIM, CDIM, 0);
    }

    // 4) WMMA neighborhood attention -> attn16 (f16)  (2304 waves, 2/block)
    na_attn_wmma<<<2304 / 2, 64, 0, stream>>>(qkvbuf, rpb, attn16);

    // 5) proj GEMM + residual(x): x2 = x + attn @ proj_w + proj_b  (576 waves)
    {
        int waves = (NPOS / 16) * (CDIM / 64);
        gemm_wmma_f16<<<(waves + 7) / 8, 256, 0, stream>>>(
            attn16, projwP, proj_b, x, x2, nullptr, NPOS, CDIM, CDIM, 0);
    }

    // 6) LN2
    ln_kernel<<<NPOS / 8, 256, 0, stream>>>(x2, norm2_g, norm2_b, h2_16, NPOS);

    // 7) FC1 GEMM + exact GELU -> m16 (f16)           (2304 waves)
    {
        int waves = (NPOS / 16) * (256 / 64);
        gemm_wmma_f16<<<(waves + 7) / 8, 256, 0, stream>>>(
            h2_16, fc1wP, fc1_b, nullptr, nullptr, m16, NPOS, 256, CDIM, 1);
    }

    // 8) FC2 GEMM + residual(x2) -> d_out (f32)       (576 waves)
    {
        int waves = (NPOS / 16) * (CDIM / 64);
        gemm_wmma_f16<<<(waves + 7) / 8, 256, 0, stream>>>(
            m16, fc2wP, fc2_b, x2, (float*)d_out, nullptr, NPOS, CDIM, 256, 0);
    }
}
